// MultiHeadAttention_11020886082027
// MI455X (gfx1250) — matrix operand order repaired and checked by reading, not compiled, not run on hardware
//
#include <hip/hip_runtime.h>
#include <hip/hip_bf16.h>

#define SEQ     2048
#define DMODEL  1024
#define NHEADS  16
#define DHEAD   64
#define BATCH   2
#define NROWS   (BATCH * SEQ)      // 4096
#define LORA_SCALE 2.0f            // ALPHA/RANK = 16/8
#define EPS_VAL (-1.0e9f)
#define INV_SQRT_D 0.125f          // 1/sqrt(64)

// Toggle for gfx1250 async memory->LDS staging (GLOBAL_LOAD_ASYNC_TO_LDS_B128).
#define USE_ASYNC_LDS 1

typedef __attribute__((ext_vector_type(16))) __bf16 bf16x16;
typedef __attribute__((ext_vector_type(8)))  __bf16 bf16x8;
typedef __attribute__((ext_vector_type(8)))  float  f32x8;

union AFrag { bf16x16 v; __bf16 e[16]; bf16x8 h[2]; };
union CFrag { f32x8 v; float e[8]; };

// ---- WMMA wrapper: D = A(16x32 bf16) * B(32x16 bf16) + C(f32) -------------
__device__ inline f32x8 wmma_bf16(bf16x16 a, bf16x16 b, f32x8 c) {
    return __builtin_amdgcn_wmma_f32_16x16x32_bf16(
        /*neg_a=*/false, a, /*neg_b=*/false, b,
        /*c_mod=*/(short)0, c, /*reuse_a=*/false, /*reuse_b=*/false);
}

// A fragment: rows row0..row0+15, K = k0..k0+31, from row-major [.., ld]
__device__ inline bf16x16 frag_a(const __bf16* base, int ld, int row0, int k0) {
    const int lane = threadIdx.x & 31;
    const __bf16* p = base + (size_t)(row0 + (lane & 15)) * ld + k0 + ((lane >> 4) << 3);
    AFrag f;
    f.h[0] = *(const bf16x8*)(p);
    f.h[1] = *(const bf16x8*)(p + 16);
    return f.v;
}

// B fragment from B^T stored row-major [N][K]: col n0+lane%16, K = 8*half+0..7 and 16+8*half+0..7
__device__ inline bf16x16 frag_b(const __bf16* baseT, int ld, int n0, int k0) {
    const int lane = threadIdx.x & 31;
    const __bf16* p = baseT + (size_t)(n0 + (lane & 15)) * ld + k0 + ((lane >> 4) << 3);
    AFrag f;
    f.h[0] = *(const bf16x8*)(p);
    f.h[1] = *(const bf16x8*)(p + 16);
    return f.v;
}

// ---- Async global -> LDS copy (16B per lane), ASYNCcnt tracked -------------
__device__ inline void g2lds_b128(const __bf16* g, __bf16* l) {
#if USE_ASYNC_LDS
    // Low 32 bits of a flat LDS pointer are the LDS byte offset (aperture rule).
    unsigned lds_off = (unsigned)(unsigned long long)l;
    asm volatile("global_load_async_to_lds_b128 %0, %1, off"
                 :: "v"(lds_off), "v"(g) : "memory");
#else
    *(bf16x8*)l = *(const bf16x8*)g;
#endif
}
__device__ inline void lds_stage_fence() {
#if USE_ASYNC_LDS
    asm volatile("s_wait_asynccnt 0" ::: "memory");
#endif
}

// ---------------------------------------------------------------------------
// Kernel 1: cast x (f32) -> bf16
__global__ void cast_f32_bf16_kernel(const float* __restrict__ in,
                                     __bf16* __restrict__ out, int n) {
    int i = blockIdx.x * blockDim.x + threadIdx.x;
    if (i < n) out[i] = (__bf16)in[i];
}

// Kernel 2: fuse LoRA into dense weight, store TRANSPOSED bf16:
// WT[n][k] = W[k][n] + SCALE * sum_r lA[k][r]*lB[r][n]
__global__ void fuse_weights_kernel(const float* __restrict__ W,
                                    const float* __restrict__ lA,
                                    const float* __restrict__ lB,
                                    __bf16* __restrict__ WT, int N) {
    const int n = blockIdx.x;
    __shared__ float lBn[8];
    if (threadIdx.x < 8) lBn[threadIdx.x] = lB[threadIdx.x * N + n];
    __syncthreads();
    for (int k = threadIdx.x; k < DMODEL; k += blockDim.x) {
        float acc = W[(size_t)k * N + n];
        float l = 0.f;
        #pragma unroll
        for (int r = 0; r < 8; ++r) l += lA[k * 8 + r] * lBn[r];
        WT[(size_t)n * DMODEL + k] = (__bf16)(acc + l * LORA_SCALE);
    }
}

// ---------------------------------------------------------------------------
// Tiled bf16 WMMA GEMM:  C[m][n] = sum_k A[m][k]*BT[n][k] + bias[n]
// Block tile 64(M) x 128(N), 8 waves in 2x4, each wave 32x32 (2x2 frags).
// K staged in 64-wide slabs (8 WMMA per barrier pair).
// MODE 0: QKV epilogue (scatter to Q [B,H,S,d], K [B,H,S,d], V^T [B,H,d,S], bf16)
// MODE 1: plain f32 output [M,N]
template <int MODE>
__global__ void __launch_bounds__(256)
gemm_bf16_kernel(const __bf16* __restrict__ A, const __bf16* __restrict__ BT,
                 const float* __restrict__ bias,
                 __bf16* __restrict__ Qp, __bf16* __restrict__ Kp,
                 __bf16* __restrict__ VTp, float* __restrict__ outp) {
    __shared__ __align__(16) __bf16 Ash[64 * 64];    //  8 KB
    __shared__ __align__(16) __bf16 Bsh[128 * 64];   // 16 KB
    const int tid  = threadIdx.x;
    const int wave = tid >> 5;
    const int wm = wave >> 2, wn = wave & 3;         // 2 x 4 wave grid
    const int m0  = blockIdx.y * 64;
    const int n0b = blockIdx.x * 128;

    CFrag c[2][2];
    #pragma unroll
    for (int i = 0; i < 2; ++i)
        #pragma unroll
        for (int jb = 0; jb < 2; ++jb)
            #pragma unroll
            for (int j = 0; j < 8; ++j) c[i][jb].e[j] = 0.f;

    for (int k0 = 0; k0 < DMODEL; k0 += 64) {
        // stage A 64x64 (two 16B chunks per thread)
        #pragma unroll
        for (int q = 0; q < 2; ++q) {
            int id = tid + q * 256;
            int r = id >> 3, c8 = (id & 7) * 8;
            g2lds_b128(&A[(size_t)(m0 + r) * DMODEL + k0 + c8], &Ash[r * 64 + c8]);
        }
        // stage B^T 128x64 (four 16B chunks per thread)
        #pragma unroll
        for (int q = 0; q < 4; ++q) {
            int id = tid + q * 256;
            int n = id >> 3, c8 = (id & 7) * 8;
            g2lds_b128(&BT[(size_t)(n0b + n) * DMODEL + k0 + c8], &Bsh[n * 64 + c8]);
        }
        if (k0 + 64 < DMODEL) {   // speculative prefetch of next slab
            __builtin_prefetch(&A[(size_t)(m0 + (tid >> 2)) * DMODEL + k0 + 64], 0, 0);
            __builtin_prefetch(&BT[(size_t)(n0b + (tid >> 1)) * DMODEL + k0 + 64], 0, 0);
        }
        lds_stage_fence();
        __syncthreads();

        #pragma unroll
        for (int kk = 0; kk < 64; kk += 32) {
            bf16x16 a0 = frag_a(Ash, 64, wm * 32, kk);
            bf16x16 a1 = frag_a(Ash, 64, wm * 32 + 16, kk);
            bf16x16 b0 = frag_b(Bsh, 64, wn * 32, kk);
            bf16x16 b1 = frag_b(Bsh, 64, wn * 32 + 16, kk);
            c[0][0].v = wmma_bf16(a0, b0, c[0][0].v);
            c[0][1].v = wmma_bf16(a0, b1, c[0][1].v);
            c[1][0].v = wmma_bf16(a1, b0, c[1][0].v);
            c[1][1].v = wmma_bf16(a1, b1, c[1][1].v);
        }
        __syncthreads();
    }

    const int lane = tid & 31, ln = lane & 15, half = lane >> 4;
    #pragma unroll
    for (int i = 0; i < 2; ++i)
        #pragma unroll
        for (int jb = 0; jb < 2; ++jb)
            #pragma unroll
            for (int j = 0; j < 8; ++j) {
                int m = m0 + wm * 32 + i * 16 + j + 8 * half;
                int n = n0b + wn * 32 + jb * 16 + ln;
                float val = c[i][jb].e[j] + bias[n];
                if (MODE == 0) {
                    int bb = m >> 11, s = m & (SEQ - 1);
                    int region = n >> 10;          // 0=q 1=k 2=v
                    int nn = n & (DMODEL - 1);
                    int h = nn >> 6, dh = nn & 63;
                    size_t hb = (size_t)(bb * NHEADS + h);
                    if (region == 0)
                        Qp[(hb * SEQ + s) * DHEAD + dh] = (__bf16)val;
                    else if (region == 1)
                        Kp[(hb * SEQ + s) * DHEAD + dh] = (__bf16)val;
                    else
                        VTp[(hb * DHEAD + dh) * SEQ + s] = (__bf16)val;
                } else {
                    outp[(size_t)m * DMODEL + n] = val;
                }
            }
}

// ---------------------------------------------------------------------------
// Flash-style causal+padded attention. Block = 8 waves, covers 128 q rows of
// one (b,h); each wave owns a 16-row q tile. K/V staged per 32-row kv tile.
__global__ void __launch_bounds__(256)
attn_kernel(const __bf16* __restrict__ Q, const __bf16* __restrict__ K,
            const __bf16* __restrict__ VT, const int* __restrict__ amask,
            __bf16* __restrict__ ctx) {
    __shared__ __align__(16) __bf16 Ksh[32 * 64];   // kv rows x d
    __shared__ __align__(16) __bf16 Vtsh[64 * 32];  // d x kv rows (V^T)
    __shared__ __align__(16) __bf16 Psh[8][16 * 32];

    const int bh = blockIdx.x;
    const int bb = bh / NHEADS, h = bh % NHEADS;
    const int tid = threadIdx.x, wave = tid >> 5, lane = tid & 31;
    const int ln = lane & 15, half = lane >> 4;
    const int qt0 = blockIdx.y * 128;
    const int qw0 = qt0 + wave * 16;

    const __bf16* Qh = Q  + (size_t)bh * SEQ * DHEAD;
    const __bf16* Kh = K  + (size_t)bh * SEQ * DHEAD;
    const __bf16* Vh = VT + (size_t)bh * DHEAD * SEQ;

    // Q fragments for this wave's 16 rows (d = 0..31, 32..63), reused all tiles
    bf16x16 qa0 = frag_a(Qh, DHEAD, qw0, 0);
    bf16x16 qa1 = frag_a(Qh, DHEAD, qw0, 32);

    CFrag o[4];
    float m_row[8], l_row[8];
    #pragma unroll
    for (int j = 0; j < 8; ++j) { m_row[j] = -3.0e38f; l_row[j] = 0.f; }
    #pragma unroll
    for (int f = 0; f < 4; ++f)
        #pragma unroll
        for (int j = 0; j < 8; ++j) o[f].e[j] = 0.f;

    const int ntiles = blockIdx.y * 4 + 4;   // causal: kv rows <= qt0+127
    for (int t = 0; t < ntiles; ++t) {
        // cooperative stage: K tile 32x64, V^T tile 64x32 (8 bf16 per thread each)
        {
            int idx = tid * 8;
            int r = idx >> 6, c8 = idx & 63;
            g2lds_b128(&Kh[(size_t)(t * 32 + r) * DHEAD + c8], &Ksh[idx]);
        }
        {
            int idx = tid * 8;
            int dh = idx >> 5, c8 = idx & 31;
            g2lds_b128(&Vh[(size_t)dh * SEQ + t * 32 + c8], &Vtsh[idx]);
        }
        lds_stage_fence();
        __syncthreads();

        if (t * 32 <= qw0 + 15) {      // tile intersects this wave's causal range
            // S = Q K^T : 16x32 tile as two 16x16 C fragments
            CFrag sc[2];
            #pragma unroll
            for (int s2 = 0; s2 < 2; ++s2)
                #pragma unroll
                for (int j = 0; j < 8; ++j) sc[s2].e[j] = 0.f;

            sc[0].v = wmma_bf16(qa0, frag_b(Ksh, 64, 0, 0),  sc[0].v);
            sc[0].v = wmma_bf16(qa1, frag_b(Ksh, 64, 0, 32), sc[0].v);
            sc[1].v = wmma_bf16(qa0, frag_b(Ksh, 64, 16, 0),  sc[1].v);
            sc[1].v = wmma_bf16(qa1, frag_b(Ksh, 64, 16, 32), sc[1].v);

            const int col0 = t * 32 + ln, col1 = col0 + 16;
            const bool pad0 = (amask[bb * SEQ + col0] == 0);
            const bool pad1 = (amask[bb * SEQ + col1] == 0);

            #pragma unroll
            for (int j = 0; j < 8; ++j) {
                const int qrow = qw0 + j + 8 * half;
                float s0 = sc[0].e[j];
                float s1 = sc[1].e[j];
                if (pad0 || col0 > qrow) s0 = EPS_VAL;   // mask BEFORE scaling (ref)
                if (pad1 || col1 > qrow) s1 = EPS_VAL;
                s0 *= INV_SQRT_D;
                s1 *= INV_SQRT_D;

                // row max across the 16 lanes holding this row
                float mt = fmaxf(s0, s1);
                #pragma unroll
                for (int off = 8; off; off >>= 1)
                    mt = fmaxf(mt, __shfl_xor(mt, off, 32));

                const float mnew = fmaxf(m_row[j], mt);
                const float corr = __expf(m_row[j] - mnew);
                const float p0 = __expf(s0 - mnew);
                const float p1 = __expf(s1 - mnew);
                float rs = p0 + p1;
                #pragma unroll
                for (int off = 8; off; off >>= 1)
                    rs += __shfl_xor(rs, off, 32);

                l_row[j] = l_row[j] * corr + rs;
                m_row[j] = mnew;
                #pragma unroll
                for (int f = 0; f < 4; ++f) o[f].e[j] *= corr;

                // spill P to LDS (C-frag layout -> row-major 16x32 bf16)
                Psh[wave][(j + 8 * half) * 32 + ln]      = (__bf16)p0;
                Psh[wave][(j + 8 * half) * 32 + ln + 16] = (__bf16)p1;
            }

            asm volatile("s_wait_dscnt 0" ::: "memory");

            // O += P * V : reload P as A fragment (16x32), V^T gives contiguous B frags
            bf16x16 pa = frag_a(&Psh[wave][0], 32, 0, 0);
            #pragma unroll
            for (int f = 0; f < 4; ++f)
                o[f].v = wmma_bf16(pa, frag_b(Vtsh, 32, f * 16, 0), o[f].v);
        }
        __syncthreads();
    }

    // ctx[b, qrow, h*64 + dcol] = O / l   (bf16, feeds out-projection GEMM)
    #pragma unroll
    for (int f = 0; f < 4; ++f)
        #pragma unroll
        for (int j = 0; j < 8; ++j) {
            int qrow = qw0 + j + 8 * half;
            float v = o[f].e[j] / l_row[j];
            ctx[((size_t)(bb * SEQ + qrow)) * DMODEL + h * DHEAD + f * 16 + ln] =
                (__bf16)v;
        }
}

// ---------------------------------------------------------------------------
extern "C" void kernel_launch(void* const* d_in, const int* in_sizes, int n_in,
                              void* d_out, int out_size, void* d_ws, size_t ws_size,
                              hipStream_t stream) {
    const float* x      = (const float*)d_in[0];
    const int*   amask  = (const int*)d_in[1];
    const float* W_qkv  = (const float*)d_in[2];
    const float* b_qkv  = (const float*)d_in[3];
    const float* lA_qkv = (const float*)d_in[4];
    const float* lB_qkv = (const float*)d_in[5];
    const float* W_out  = (const float*)d_in[6];
    const float* b_out  = (const float*)d_in[7];
    const float* lA_out = (const float*)d_in[8];
    const float* lB_out = (const float*)d_in[9];
    float* out = (float*)d_out;

    char* ws = (char*)d_ws;
    size_t off = 0;
    auto wsalloc = [&](size_t bytes) {
        void* p = ws + off;
        off = (off + bytes + 255) & ~(size_t)255;
        return p;
    };
    __bf16* xb   = (__bf16*)wsalloc((size_t)NROWS * DMODEL * 2);       // 8 MB
    __bf16* WqT  = (__bf16*)wsalloc((size_t)3 * DMODEL * DMODEL * 2);  // 6 MB
    __bf16* WoT  = (__bf16*)wsalloc((size_t)DMODEL * DMODEL * 2);      // 2 MB
    __bf16* Qb   = (__bf16*)wsalloc((size_t)NROWS * DMODEL * 2);       // 8 MB
    __bf16* Kb   = (__bf16*)wsalloc((size_t)NROWS * DMODEL * 2);       // 8 MB
    __bf16* VTb  = (__bf16*)wsalloc((size_t)NROWS * DMODEL * 2);       // 8 MB
    __bf16* ctxb = (__bf16*)wsalloc((size_t)NROWS * DMODEL * 2);       // 8 MB

    // 1) cast x -> bf16
    cast_f32_bf16_kernel<<<(NROWS * DMODEL) / 256, 256, 0, stream>>>(
        x, xb, NROWS * DMODEL);

    // 2) fold LoRA into dense weights, transpose + cast to bf16
    fuse_weights_kernel<<<3 * DMODEL, 256, 0, stream>>>(W_qkv, lA_qkv, lB_qkv,
                                                        WqT, 3 * DMODEL);
    fuse_weights_kernel<<<DMODEL, 256, 0, stream>>>(W_out, lA_out, lB_out,
                                                    WoT, DMODEL);

    // 3) QKV GEMM (+bias) with head-layout scatter epilogue
    gemm_bf16_kernel<0><<<dim3(3 * DMODEL / 128, NROWS / 64), 256, 0, stream>>>(
        xb, WqT, b_qkv, Qb, Kb, VTb, nullptr);

    // 4) flash attention (causal + pad mask, per reference semantics)
    attn_kernel<<<dim3(BATCH * NHEADS, SEQ / 128), 256, 0, stream>>>(
        Qb, Kb, VTb, amask, ctxb);

    // 5) output projection (+bias) -> f32 result
    gemm_bf16_kernel<1><<<dim3(DMODEL / 128, NROWS / 64), 256, 0, stream>>>(
        ctxb, WoT, b_out, nullptr, nullptr, nullptr, out);
}
